// RWKVXBlock_46712064311934
// MI455X (gfx1250) — matrix operand order repaired and checked by reading, not compiled, not run on hardware
//
#include <hip/hip_runtime.h>
#include <hip/hip_bf16.h>
#include <math.h>

// ---------------------------------------------------------------------------
// Types for CDNA5 WMMA (wave32): V_WMMA_F32_16X16X32_BF16
// ---------------------------------------------------------------------------
typedef __attribute__((ext_vector_type(16))) __bf16 bf16x16;
typedef __attribute__((ext_vector_type(8)))  __bf16 bf16x8;
typedef __attribute__((ext_vector_type(8)))  float  f32x8;

#define WMMA_BF16(A, B, C) \
  __builtin_amdgcn_wmma_f32_16x16x32_bf16(false, (A), false, (B), (short)0, (C), false, false)

// float -> bf16 round-to-nearest-even via bit ops
__device__ __forceinline__ __bf16 f2bf(float f) {
  union { float f; unsigned int u; } in; in.f = f;
  unsigned int x = in.u;
  unsigned int r = x + 0x7fffu + ((x >> 16) & 1u);
  union { unsigned short u; __bf16 b; } out;
  out.u = (unsigned short)(r >> 16);
  return out.b;
}

__device__ __forceinline__ float sigmoidf_(float x) {
  return 1.0f / (1.0f + __expf(-x));
}

// ---------------------------------------------------------------------------
// Problem constants
// ---------------------------------------------------------------------------
#define DIM   1024
#define HEADA 64
#define BATCH 2
#define SEQ   4096
#define NTOK  (BATCH * SEQ)     // 8192
#define DI    4096
#define TOPK  32
#define CHUNK 256               // attention key chunk

// ---------------------------------------------------------------------------
// f32 -> bf16 weight conversion
// ---------------------------------------------------------------------------
__global__ __launch_bounds__(256) void cvt_f32_bf16_kernel(const float* __restrict__ src,
                                                           __bf16* __restrict__ dst, int n) {
  int i = blockIdx.x * 256 + threadIdx.x;
  if (i < n) dst[i] = f2bf(src[i]);
}

// ---------------------------------------------------------------------------
// RMSNorm: one block per token
// ---------------------------------------------------------------------------
__global__ __launch_bounds__(256) void rmsnorm_kernel(const float* __restrict__ x,
                                                      const float* __restrict__ w,
                                                      float* __restrict__ out) {
  __shared__ float red[256];
  const size_t base = (size_t)blockIdx.x * DIM;
  int tid = threadIdx.x;
  float ss = 0.f;
  for (int d = tid; d < DIM; d += 256) { float v = x[base + d]; ss += v * v; }
  red[tid] = ss; __syncthreads();
  for (int s = 128; s > 0; s >>= 1) {
    if (tid < s) red[tid] += red[tid + s];
    __syncthreads();
  }
  float norm = sqrtf(red[0]) * (1.0f / 32.0f);   // ||x|| / sqrt(1024)
  float inv = 1.0f / (norm + 1e-8f);
  for (int d = tid; d < DIM; d += 256) out[base + d] = w[d] * x[base + d] * inv;
}

// ---------------------------------------------------------------------------
// time-mix prep: token shift + mixes + bf16 conversion
// ---------------------------------------------------------------------------
__global__ __launch_bounds__(256) void tmix_prep_kernel(const float* __restrict__ xn,
                                                        const float* __restrict__ mv,
                                                        const float* __restrict__ mr,
                                                        __bf16* __restrict__ mvb,
                                                        __bf16* __restrict__ mrb,
                                                        __bf16* __restrict__ xnb) {
  size_t idx = (size_t)blockIdx.x * 256 + threadIdx.x;
  if (idx >= (size_t)NTOK * DIM) return;
  size_t t = idx / DIM; int d = (int)(idx % DIM);
  int s = (int)(t % SEQ);
  size_t pidx = (s == 0) ? idx : idx - DIM;
  float xv = xn[idx], xp = xn[pidx];
  float a = mv[d], b = mr[d];
  mvb[idx] = f2bf(xv * a + xp * (1.0f - a));
  mrb[idx] = f2bf(xv * b + xp * (1.0f - b));
  xnb[idx] = f2bf(xv);
}

// channel-mix prep: two outputs
__global__ __launch_bounds__(256) void cmix_prep_kernel(const float* __restrict__ xn,
                                                        const float* __restrict__ mk,
                                                        const float* __restrict__ mr,
                                                        __bf16* __restrict__ mkb,
                                                        __bf16* __restrict__ mrb) {
  size_t idx = (size_t)blockIdx.x * 256 + threadIdx.x;
  if (idx >= (size_t)NTOK * DIM) return;
  size_t t = idx / DIM; int d = (int)(idx % DIM);
  int s = (int)(t % SEQ);
  size_t pidx = (s == 0) ? idx : idx - DIM;
  float xv = xn[idx], xp = xn[pidx];
  float a = mk[d], b = mr[d];
  mkb[idx] = f2bf(xv * a + xp * (1.0f - a));
  mrb[idx] = f2bf(xv * b + xp * (1.0f - b));
}

// rkv = bf16( sigmoid(r) * (aout + v) )
__global__ __launch_bounds__(256) void rkv_prep_kernel(const float* __restrict__ r,
                                                       const float* __restrict__ aout,
                                                       const float* __restrict__ v,
                                                       __bf16* __restrict__ rkvb) {
  size_t idx = (size_t)blockIdx.x * 256 + threadIdx.x;
  if (idx >= (size_t)NTOK * DIM) return;
  rkvb[idx] = f2bf(sigmoidf_(r[idx]) * (aout[idx] + v[idx]));
}

// out = x1 + sigmoid(r2) * kv
__global__ __launch_bounds__(256) void final_kernel(const float* __restrict__ x1,
                                                    const float* __restrict__ r2,
                                                    const float* __restrict__ kv,
                                                    float* __restrict__ out) {
  size_t idx = (size_t)blockIdx.x * 256 + threadIdx.x;
  if (idx >= (size_t)NTOK * DIM) return;
  out[idx] = x1[idx] + sigmoidf_(r2[idx]) * kv[idx];
}

// ---------------------------------------------------------------------------
// Generic WMMA GEMM:  out[m,n] = sum_k A[m,k] * W[n,k]  (+bias[n]) (+epilogue)
// A: (M,K) bf16 row-major.  W: (N,K) bf16 row-major (x @ W.T pattern).
// Block = 128 threads = 4 waves; block tile 128(M) x 64(N);
// each wave: 32(M) x 64(N) -> 2 A fragments x 4 B fragments = 8 wmma / k-step,
// 12 b128 loads per k-step (B fragments shared across both A rows).
// mode: 0 = f32 out; 1 = bf16 out; 2 = relu(v)^2 -> bf16; 3 = f32 + residual
// ---------------------------------------------------------------------------
__global__ __launch_bounds__(128) void gemm_bf16_kernel(const __bf16* __restrict__ A,
                                                        const __bf16* __restrict__ W,
                                                        const float* __restrict__ bias,
                                                        const float* __restrict__ residual,
                                                        float* __restrict__ outF,
                                                        __bf16* __restrict__ outB,
                                                        int M, int N, int K, int mode) {
  const int lane  = threadIdx.x & 31;
  const int wave  = threadIdx.x >> 5;
  const int row16 = lane & 15;
  const int hi    = lane >> 4;

  const int m0 = blockIdx.y * 128 + wave * 32;
  const int n0 = blockIdx.x * 64;

  const __bf16* arow0 = A + (size_t)(m0 + row16) * K;        // rows m0..m0+15
  const __bf16* arow1 = A + (size_t)(m0 + 16 + row16) * K;   // rows m0+16..m0+31

  f32x8 acc[2][4] = {};

  for (int k0 = 0; k0 < K; k0 += 32) {
    // A fragments (16x32 bf16 each): lanes 0-15 -> K {k0..k0+15},
    // lanes 16-31 -> K {k0+16..k0+31}
    bf16x8 a0lo = *(const bf16x8*)(arow0 + k0 + hi * 16);
    bf16x8 a0hi = *(const bf16x8*)(arow0 + k0 + 8 + hi * 16);
    bf16x16 af0 = __builtin_shufflevector(a0lo, a0hi, 0, 1, 2, 3, 4, 5, 6, 7,
                                          8, 9, 10, 11, 12, 13, 14, 15);
    bf16x8 a1lo = *(const bf16x8*)(arow1 + k0 + hi * 16);
    bf16x8 a1hi = *(const bf16x8*)(arow1 + k0 + 8 + hi * 16);
    bf16x16 af1 = __builtin_shufflevector(a1lo, a1hi, 0, 1, 2, 3, 4, 5, 6, 7,
                                          8, 9, 10, 11, 12, 13, 14, 15);
#pragma unroll
    for (int t = 0; t < 4; ++t) {
      // B fragment (32x16): lane holds column n = n0+t*16+row16,
      // K {k0..k0+15} (lanes 0-15) or {k0+16..k0+31} (lanes 16-31)
      const __bf16* wrow = W + (size_t)(n0 + t * 16 + row16) * K + k0 + hi * 16;
      bf16x16 bfrag = *(const bf16x16*)wrow;
      acc[0][t] = WMMA_BF16(af0, bfrag, acc[0][t]);
      acc[1][t] = WMMA_BF16(af1, bfrag, acc[1][t]);
    }
  }

#pragma unroll
  for (int h = 0; h < 2; ++h) {
#pragma unroll
    for (int t = 0; t < 4; ++t) {
      const int n = n0 + t * 16 + row16;
      const float bi = bias ? bias[n] : 0.0f;
#pragma unroll
      for (int j = 0; j < 8; ++j) {
        const int m = m0 + h * 16 + j + 8 * hi;  // D layout: vgpr j -> M=j (+8 for hi lanes)
        const size_t idx = (size_t)m * N + n;
        float v = acc[h][t][j] + bi;
        if (mode == 0) {
          outF[idx] = v;
        } else if (mode == 1) {
          outB[idx] = f2bf(v);
        } else if (mode == 2) {
          float rl = v > 0.0f ? v : 0.0f;
          outB[idx] = f2bf(rl * rl);
        } else {
          outF[idx] = v + residual[idx];
        }
      }
    }
  }
}

// ---------------------------------------------------------------------------
// Sparse attention: one block = 16 query rows of one batch.
// WMMA score tiles (Q 16x64 bf16 x K^T) into LDS, streaming exact top-32 per
// row (replace-min), softmax over survivors, gather-accumulate 32 V rows.
// Output written as bf16 (feeds the sa_o_w GEMM).
// ---------------------------------------------------------------------------
__global__ __launch_bounds__(256) void sparse_attn_kernel(const __bf16* __restrict__ Qb,
                                                          const __bf16* __restrict__ Kb,
                                                          const float* __restrict__ Vsa,
                                                          __bf16* __restrict__ outB) {
  __shared__ float sc[16][CHUNK];
  __shared__ float tops[16][TOPK];
  __shared__ int   topi[16][TOPK];

  const int tid  = threadIdx.x;
  const int lane = tid & 31;
  const int wave = tid >> 5;               // 0..7
  const int row16 = lane & 15;
  const int hi    = lane >> 4;

  const int b  = blockIdx.y;
  const int q0 = blockIdx.x * 16;

  const __bf16* Qbase = Qb + (size_t)b * SEQ * HEADA;
  const __bf16* Kbase = Kb + (size_t)b * SEQ * HEADA;
  const float*  Vbase = Vsa + (size_t)b * SEQ * DIM;

  // Per-wave Q fragments (reused across all key chunks): head dim 64 = 2 k-steps
  const __bf16* qrow = Qbase + (size_t)(q0 + row16) * HEADA;
  bf16x8 l0 = *(const bf16x8*)(qrow + hi * 16);
  bf16x8 h0 = *(const bf16x8*)(qrow + 8 + hi * 16);
  bf16x16 aq0 = __builtin_shufflevector(l0, h0, 0, 1, 2, 3, 4, 5, 6, 7,
                                        8, 9, 10, 11, 12, 13, 14, 15);
  bf16x8 l1 = *(const bf16x8*)(qrow + 32 + hi * 16);
  bf16x8 h1 = *(const bf16x8*)(qrow + 40 + hi * 16);
  bf16x16 aq1 = __builtin_shufflevector(l1, h1, 0, 1, 2, 3, 4, 5, 6, 7,
                                        8, 9, 10, 11, 12, 13, 14, 15);

  // init top-k lists
  for (int i = tid; i < 16 * TOPK; i += 256) {
    tops[i / TOPK][i % TOPK] = -__builtin_inff();
    topi[i / TOPK][i % TOPK] = 0;
  }
  __syncthreads();

  float minv = -__builtin_inff();   // live only in threads 0..15
  int   minp = 0;

  for (int ch = 0; ch < SEQ / CHUNK; ++ch) {
#pragma unroll
    for (int sub = 0; sub < 2; ++sub) {
      const int kt  = wave + sub * 8;          // 16 key tiles of 16 per chunk
      const int kb0 = ch * CHUNK + kt * 16;
      const __bf16* krow = Kbase + (size_t)(kb0 + row16) * HEADA;
      bf16x16 b0 = *(const bf16x16*)(krow + hi * 16);        // K 0..31
      bf16x16 b1 = *(const bf16x16*)(krow + 32 + hi * 16);   // K 32..63
      f32x8 acc = {};
      acc = WMMA_BF16(aq0, b0, acc);
      acc = WMMA_BF16(aq1, b1, acc);
#pragma unroll
      for (int j = 0; j < 8; ++j)
        sc[j + 8 * hi][kt * 16 + row16] = acc[j] * 0.125f;   // 1/sqrt(64)
    }
    __syncthreads();

    // streaming exact top-32 (replace-min), one thread per query row
    if (tid < 16) {
      for (int c = 0; c < CHUNK; ++c) {
        float v = sc[tid][c];
        if (v > minv) {
          tops[tid][minp] = v;
          topi[tid][minp] = ch * CHUNK + c;
          float mv2 = tops[tid][0]; int mp2 = 0;
          for (int j = 1; j < TOPK; ++j) {
            float tv = tops[tid][j];
            if (tv < mv2) { mv2 = tv; mp2 = j; }
          }
          minv = mv2; minp = mp2;
        }
      }
    }
    __syncthreads();
  }

  // softmax over the 32 survivors (weights written back into tops)
  if (tid < 16) {
    float mx = tops[tid][0];
    for (int j = 1; j < TOPK; ++j) mx = fmaxf(mx, tops[tid][j]);
    float s = 0.f;
    for (int j = 0; j < TOPK; ++j) { float e = __expf(tops[tid][j] - mx); tops[tid][j] = e; s += e; }
    float inv = 1.0f / s;
    for (int j = 0; j < TOPK; ++j) tops[tid][j] *= inv;
  }
  __syncthreads();

  // out[q0+r, :] = sum_j w_j * V[idx_j, :]   (coalesced gather over channels)
  for (int r = 0; r < 16; ++r) {
    float wv[TOPK]; int iv[TOPK];
#pragma unroll
    for (int j = 0; j < TOPK; ++j) { wv[j] = tops[r][j]; iv[j] = topi[r][j]; }
    __bf16* orow = outB + ((size_t)b * SEQ + q0 + r) * DIM;
    for (int d = tid; d < DIM; d += 256) {
      float acc = 0.f;
#pragma unroll
      for (int j = 0; j < TOPK; ++j) acc += wv[j] * Vbase[(size_t)iv[j] * DIM + d];
      orow[d] = f2bf(acc);
    }
  }
}

// ---------------------------------------------------------------------------
// Host launcher
// ---------------------------------------------------------------------------
static inline void launch_gemm(const __bf16* A, const __bf16* W, const float* bias,
                               const float* residual, float* outF, __bf16* outB,
                               int M, int N, int K, int mode, hipStream_t stream) {
  dim3 grid(N / 64, M / 128);
  gemm_bf16_kernel<<<grid, 128, 0, stream>>>(A, W, bias, residual, outF, outB, M, N, K, mode);
}

extern "C" void kernel_launch(void* const* d_in, const int* in_sizes, int n_in,
                              void* d_out, int out_size, void* d_ws, size_t ws_size,
                              hipStream_t stream) {
  (void)in_sizes; (void)n_in; (void)out_size; (void)ws_size;

  const float* x         = (const float*)d_in[0];
  const float* norm1_w   = (const float*)d_in[1];
  // d_in[2] tm_mix_k and d_in[5] tm_key_w are dead code in the reference
  const float* tm_mix_v  = (const float*)d_in[3];
  const float* tm_mix_r  = (const float*)d_in[4];
  const float* tm_value_w  = (const float*)d_in[6];
  const float* tm_recept_w = (const float*)d_in[7];
  const float* tm_out_w    = (const float*)d_in[8];
  const float* sa_q_w  = (const float*)d_in[9];
  const float* sa_q_b  = (const float*)d_in[10];
  const float* sa_k_w  = (const float*)d_in[11];
  const float* sa_k_b  = (const float*)d_in[12];
  const float* sa_v_w  = (const float*)d_in[13];
  const float* sa_v_b  = (const float*)d_in[14];
  const float* sa_o_w  = (const float*)d_in[15];
  const float* sa_o_b  = (const float*)d_in[16];
  const float* norm2_w = (const float*)d_in[17];
  const float* cm_mix_k = (const float*)d_in[18];
  const float* cm_mix_r = (const float*)d_in[19];
  const float* cm_key_w    = (const float*)d_in[20];
  const float* cm_recept_w = (const float*)d_in[21];
  const float* cm_value_w  = (const float*)d_in[22];
  float* out = (float*)d_out;

  char* ws = (char*)d_ws;
  const size_t MB = 1ull << 20;

  // Activation buffers (lifetime-aliased bump allocation)
  float*  xn     = (float*) (ws +   0 * MB);   // 32MB; reused for rmsnorm2
  __bf16* xnb    = (__bf16*)(ws +  32 * MB);   // 16MB; later cm k-mix input
  __bf16* mvb    = (__bf16*)(ws +  48 * MB);   // 16MB; later cm r-mix input
  __bf16* mrb    = (__bf16*)(ws +  64 * MB);   // 16MB; later rkvb
  float*  vbuf   = (float*) (ws +  80 * MB);   // 32MB; later kv
  float*  rbuf   = (float*) (ws + 112 * MB);   // 32MB; later r2
  __bf16* Qb     = (__bf16*)(ws + 144 * MB);   //  1MB
  __bf16* Kb     = (__bf16*)(ws + 146 * MB);   //  1MB
  float*  Vsa    = (float*) (ws + 148 * MB);   // 32MB
  __bf16* sattnb = (__bf16*)(ws + 180 * MB);   // 16MB
  float*  aout   = (float*) (ws + 212 * MB);   // 32MB; later x1 (aliased)
  float*  x1     = aout;                       // aout dead after rkv_prep
  __bf16* k2b    = (__bf16*)(ws + 148 * MB);   // 64MB; aliases Vsa+sattnb (dead by then)
  __bf16* cm_kb  = xnb;
  __bf16* cm_rb  = mvb;
  __bf16* rkvb   = mrb;
  float*  r2     = rbuf;
  float*  kv     = vbuf;

  // bf16 weight staging area
  __bf16* wb = (__bf16*)(ws + 244 * MB);
  __bf16* wb_tmv = wb;                 // 1024*1024
  __bf16* wb_tmr = wb_tmv + 1048576;
  __bf16* wb_tmo = wb_tmr + 1048576;
  __bf16* wb_q   = wb_tmo + 1048576;   // 64*1024
  __bf16* wb_k   = wb_q   + 65536;
  __bf16* wb_sav = wb_k   + 65536;
  __bf16* wb_sao = wb_sav + 1048576;
  __bf16* wb_cmk = wb_sao + 1048576;   // 4096*1024
  __bf16* wb_cmr = wb_cmk + 4194304;
  __bf16* wb_cmv = wb_cmr + 1048576;   // 1024*4096

  auto cvt = [&](const float* s, __bf16* d, int n) {
    cvt_f32_bf16_kernel<<<(n + 255) / 256, 256, 0, stream>>>(s, d, n);
  };
  cvt(tm_value_w,  wb_tmv, 1048576);
  cvt(tm_recept_w, wb_tmr, 1048576);
  cvt(tm_out_w,    wb_tmo, 1048576);
  cvt(sa_q_w,      wb_q,   65536);
  cvt(sa_k_w,      wb_k,   65536);
  cvt(sa_v_w,      wb_sav, 1048576);
  cvt(sa_o_w,      wb_sao, 1048576);
  cvt(cm_key_w,    wb_cmk, 4194304);
  cvt(cm_recept_w, wb_cmr, 1048576);
  cvt(cm_value_w,  wb_cmv, 4194304);

  const int nElem   = NTOK * DIM;
  const int ewGrid  = (nElem + 255) / 256;

  // ---- time mix ----
  rmsnorm_kernel<<<NTOK, 256, 0, stream>>>(x, norm1_w, xn);
  tmix_prep_kernel<<<ewGrid, 256, 0, stream>>>(xn, tm_mix_v, tm_mix_r, mvb, mrb, xnb);

  launch_gemm(mvb, wb_tmv, nullptr, nullptr, vbuf, nullptr, NTOK, DIM, DIM, 0, stream);   // v
  launch_gemm(mrb, wb_tmr, nullptr, nullptr, rbuf, nullptr, NTOK, DIM, DIM, 0, stream);   // r
  launch_gemm(xnb, wb_q,   sa_q_b, nullptr, nullptr, Qb,   NTOK, HEADA, DIM, 1, stream);  // Q (bf16)
  launch_gemm(xnb, wb_k,   sa_k_b, nullptr, nullptr, Kb,   NTOK, HEADA, DIM, 1, stream);  // K (bf16)
  launch_gemm(xnb, wb_sav, sa_v_b, nullptr, Vsa, nullptr,  NTOK, DIM, DIM, 0, stream);    // V

  sparse_attn_kernel<<<dim3(SEQ / 16, BATCH), 256, 0, stream>>>(Qb, Kb, Vsa, sattnb);

  launch_gemm(sattnb, wb_sao, sa_o_b, nullptr, aout, nullptr, NTOK, DIM, DIM, 0, stream); // o-proj

  rkv_prep_kernel<<<ewGrid, 256, 0, stream>>>(rbuf, aout, vbuf, rkvb);

  // x1 = x + rkv @ tm_out_w.T (fused residual epilogue)
  launch_gemm(rkvb, wb_tmo, nullptr, x, x1, nullptr, NTOK, DIM, DIM, 3, stream);

  // ---- channel mix ----
  rmsnorm_kernel<<<NTOK, 256, 0, stream>>>(x1, norm2_w, xn);
  cmix_prep_kernel<<<ewGrid, 256, 0, stream>>>(xn, cm_mix_k, cm_mix_r, cm_kb, cm_rb);

  launch_gemm(cm_kb, wb_cmk, nullptr, nullptr, nullptr, k2b, NTOK, DI, DIM, 2, stream);   // relu^2 -> bf16
  launch_gemm(cm_rb, wb_cmr, nullptr, nullptr, r2, nullptr,  NTOK, DIM, DIM, 0, stream);  // r
  launch_gemm(k2b,   wb_cmv, nullptr, nullptr, kv, nullptr,  NTOK, DIM, DI,  0, stream);  // kv

  final_kernel<<<ewGrid, 256, 0, stream>>>(x1, r2, kv, out);
}
